// TATransEModel_20615843021254
// MI455X (gfx1250) — matrix operand order repaired and checked by reading, not compiled, not run on hardware
//
#include <hip/hip_runtime.h>
#include <hip/hip_bf16.h>
#include <math.h>

typedef __attribute__((ext_vector_type(16))) _Float16 v16h;
typedef __attribute__((ext_vector_type(8)))  _Float16 v8h;
typedef __attribute__((ext_vector_type(8)))  float    v8f;

// Problem constants (from reference)
static constexpr int BQ    = 4096;   // batch per branch
static constexpr int TQ    = 8;      // temporal length
static constexpr int DQ    = 512;    // hidden dim
static constexpr int GQ    = 2048;   // 4*D gate dim
static constexpr int TEMQ  = 32;     // #temporal embeddings
static constexpr int M_ALL = 2 * BQ; // pos + neg folded into one GEMM M

// Branch-free fast activations: v_exp_f32 + v_rcp_f32, no libm piecewise path.
__device__ __forceinline__ float fast_sigm(float x) {
  // exp(-x) -> inf for very negative x; rcp(inf) == 0, so saturation is free.
  return __builtin_amdgcn_rcpf(1.0f + __expf(-x));
}
__device__ __forceinline__ float fast_tanh(float x) {
  float xc = fminf(fmaxf(x, -10.0f), 10.0f);   // avoid inf*0 = NaN; tanh(10)=1-4e-9
  float e  = __expf(2.0f * xc);
  return (e - 1.0f) * __builtin_amdgcn_rcpf(e + 1.0f);
}

// ---------------------------------------------------------------------------
// XW[t][col] = dot(tem_emb[t], W_ih[col]) + b_ih[col] + b_hh[col]
// Only 32 distinct LSTM inputs exist (TEM=32) -> the whole x@W_ih^T GEMM
// becomes a 32x2048 precompute (33M MACs) + per-row gather.
// ---------------------------------------------------------------------------
__global__ void xw_kernel(const float* __restrict__ tem_emb,
                          const float* __restrict__ W_ih,
                          const float* __restrict__ b_ih,
                          const float* __restrict__ b_hh,
                          float* __restrict__ XW) {
  int idx = blockIdx.x * blockDim.x + threadIdx.x;   // 0 .. 32*2048-1
  if (idx >= TEMQ * GQ) return;
  int t   = idx >> 11;       // /2048
  int col = idx & (GQ - 1);
  const float4* te = (const float4*)(tem_emb + (size_t)t * DQ);
  const float4* wi = (const float4*)(W_ih + (size_t)col * DQ);
  float s = 0.f;
#pragma unroll 4
  for (int k = 0; k < DQ / 4; k++) {
    float4 a = te[k], b = wi[k];
    s = fmaf(a.x, b.x, fmaf(a.y, b.y, fmaf(a.z, b.z, fmaf(a.w, b.w, s))));
  }
  XW[idx] = s + b_ih[col] + b_hh[col];
}

// Convert W_hh -> f16 (B operand) and zero h16 / c32 state.
__global__ void prep_kernel(const float* __restrict__ W_hh,
                            _Float16* __restrict__ Whh16,
                            _Float16* __restrict__ h16,
                            float* __restrict__ c32) {
  int idx = blockIdx.x * blockDim.x + threadIdx.x;
  if (idx < GQ * DQ) Whh16[idx] = (_Float16)W_hh[idx];
  if (idx < M_ALL * DQ) { h16[idx] = (_Float16)0.f; c32[idx] = 0.f; }
}

// ---------------------------------------------------------------------------
// One LSTM step, fully fused:
//   gates(M_ALL x 2048) = gather(XW, tem_idx) + h_src @ W_hh^T   (WMMA f16)
//   i,f,g,o -> c,h update in the GEMM epilogue (gates never touch memory)
// One wave owns a 16(M) x 32(N) tile for ALL FOUR gates (8 accumulators):
//  - one A-fragment load feeds 8 WMMAs (halves h re-reads from L2 vs 16-wide N)
//  - all 8 waves in a block share identical B addresses -> WGP$ broadcast
// ---------------------------------------------------------------------------
__global__ void __launch_bounds__(256, 1)
lstm_step_kernel(const _Float16* __restrict__ h_src,
                 _Float16* __restrict__ h_dst,
                 float* __restrict__ c32,
                 const float* __restrict__ XW,
                 const _Float16* __restrict__ Whh16,
                 const int* __restrict__ pos_tem,
                 const int* __restrict__ neg_tem,
                 int step) {
  const int lane = threadIdx.x & 31;
  const int wave = threadIdx.x >> 5;
  const int n0 = blockIdx.x * 32;              // 32 hidden units per block/wave
  const int m0 = blockIdx.y * 128 + wave * 16; // pos rows 0..4095, neg 4096..8191
  const int nl = lane & 15;
  const bool hihalf = lane >= 16;

  // --- accumulator init: gates = XW[tem_row] (x@W_ih^T + both biases) ---
  int trow[8];
#pragma unroll
  for (int r = 0; r < 8; r++) {
    int m = m0 + (hihalf ? 8 + r : r);
    trow[r] = (m < BQ) ? pos_tem[m * TQ + 3 + step]
                       : neg_tem[(m - BQ) * TQ + 3 + step];
  }
  v8f acc[4][2];   // [gate][n-subtile]
#pragma unroll
  for (int g = 0; g < 4; g++)
#pragma unroll
    for (int j = 0; j < 2; j++)
#pragma unroll
      for (int r = 0; r < 8; r++)
        acc[g][j][r] = XW[(size_t)trow[r] * GQ + g * DQ + n0 + j * 16 + nl];

  // --- K loop: gates += h_src @ W_hh^T via v_wmma_f32_16x16x32_f16 ---
  // A frag: lane holds row m0+nl, K halves {kA..kA+15}, kA = hi?16:0
  // B frag: lane holds W_hh row (gate col), K halves {kB..kB+15}, kB = hi?16:0
  const _Float16* aRow = h_src + (size_t)(m0 + nl) * DQ + (hihalf ? 16 : 0);
  const int kB = hihalf ? 16 : 0;
  const _Float16* bBase[4][2];
#pragma unroll
  for (int g = 0; g < 4; g++)
#pragma unroll
    for (int j = 0; j < 2; j++)
      bBase[g][j] = Whh16 + (size_t)(g * DQ + n0 + j * 16 + nl) * DQ + kB;

#pragma unroll 2
  for (int k = 0; k < DQ; k += 32) {
    v8h a0 = *(const v8h*)(aRow + k);
    v8h a1 = *(const v8h*)(aRow + k + 8);
    v16h a;
#pragma unroll
    for (int i = 0; i < 8; i++) { a[i] = a0[i]; a[i + 8] = a1[i]; }
#pragma unroll
    for (int g = 0; g < 4; g++) {
#pragma unroll
      for (int j = 0; j < 2; j++) {
        v16h b = *(const v16h*)(bBase[g][j] + k);
        acc[g][j] = __builtin_amdgcn_wmma_f32_16x16x32_f16(
            /*neg_a=*/false, a, /*neg_b=*/false, b,
            /*c_mod=*/(short)0, acc[g][j], /*reuse_a=*/false, /*reuse_b=*/false);
      }
    }
  }

  // --- epilogue: LSTM cell update, register-local per (m,n) ---
#pragma unroll
  for (int j = 0; j < 2; j++) {
#pragma unroll
    for (int r = 0; r < 8; r++) {
      int m = m0 + (hihalf ? 8 + r : r);
      size_t off = (size_t)m * DQ + n0 + j * 16 + nl;
      float iv = fast_sigm(acc[0][j][r]);
      float fv = fast_sigm(acc[1][j][r]);
      float gv = fast_tanh(acc[2][j][r]);
      float ov = fast_sigm(acc[3][j][r]);
      float c_new = fv * c32[off] + iv * gv;
      c32[off] = c_new;
      h_dst[off] = (_Float16)(ov * fast_tanh(c_new));
    }
  }
}

// ---------------------------------------------------------------------------
// score[b] = sum_d | ent[h_idx[b]][d] + h[b][d] - ent[t_idx[b]][d] |
// One block per row; HBM-gather bound (~33 MB total -> ~1.5us at 23.3 TB/s).
// ---------------------------------------------------------------------------
__global__ void score_kernel(const _Float16* __restrict__ h16,
                             const float* __restrict__ ent_emb,
                             const int* __restrict__ pos_h,
                             const int* __restrict__ pos_t,
                             const int* __restrict__ neg_h,
                             const int* __restrict__ neg_t,
                             float* __restrict__ out) {
  __shared__ float red[256];
  int b = blockIdx.x;  // 0..8191 (pos then neg)
  int hidx, tidx;
  if (b < BQ) { hidx = pos_h[b];      tidx = pos_t[b]; }
  else        { hidx = neg_h[b - BQ]; tidx = neg_t[b - BQ]; }
  const float*    eh = ent_emb + (size_t)hidx * DQ;
  const float*    et = ent_emb + (size_t)tidx * DQ;
  const _Float16* hh = h16 + (size_t)b * DQ;
  float s = 0.f;
  for (int d = threadIdx.x; d < DQ; d += 256)
    s += fabsf(eh[d] + (float)hh[d] - et[d]);
  red[threadIdx.x] = s;
  __syncthreads();
  for (int off = 128; off > 0; off >>= 1) {
    if (threadIdx.x < off) red[threadIdx.x] += red[threadIdx.x + off];
    __syncthreads();
  }
  if (threadIdx.x == 0) out[b] = red[0];
}

// ---------------------------------------------------------------------------
extern "C" void kernel_launch(void* const* d_in, const int* in_sizes, int n_in,
                              void* d_out, int out_size, void* d_ws, size_t ws_size,
                              hipStream_t stream) {
  (void)in_sizes; (void)n_in; (void)out_size; (void)ws_size;
  const int*   pos_h   = (const int*)d_in[0];
  const int*   pos_t   = (const int*)d_in[1];
  // d_in[2] pos_r / d_in[6] neg_r / d_in[9] rel_emb are dead: the LSTM window
  // seq[:,4:8] excludes the relation slot (seq[:,0]).
  const int*   pos_tem = (const int*)d_in[3];
  const int*   neg_h   = (const int*)d_in[4];
  const int*   neg_t   = (const int*)d_in[5];
  const int*   neg_tem = (const int*)d_in[7];
  const float* ent_emb = (const float*)d_in[8];
  const float* tem_emb = (const float*)d_in[10];
  const float* W_ih    = (const float*)d_in[11];
  const float* W_hh    = (const float*)d_in[12];
  const float* b_ih    = (const float*)d_in[13];
  const float* b_hh    = (const float*)d_in[14];

  // Workspace layout (all offsets 256B-aligned), ~34.3 MB total:
  char* ws = (char*)d_ws;
  float*    XW    = (float*)(ws);                 //   256 KB  (32 x 2048 f32)
  _Float16* WHH16 = (_Float16*)(ws + 262144);     //     2 MB  (2048 x 512 f16)
  _Float16* H16A  = (_Float16*)(ws + 2359296);    //     8 MB  (8192 x 512 f16)
  _Float16* H16B  = (_Float16*)(ws + 10747904);   //     8 MB  (double buffer)
  float*    C32   = (float*)(ws + 19136512);      //    16 MB  (8192 x 512 f32)

  xw_kernel<<<(TEMQ * GQ) / 256, 256, 0, stream>>>(tem_emb, W_ih, b_ih, b_hh, XW);
  prep_kernel<<<(M_ALL * DQ) / 256, 256, 0, stream>>>(W_hh, WHH16, H16A, C32);

  dim3 grid(DQ / 32, M_ALL / 128);  // 16 n-blocks x 64 m-blocks, 8 waves/block
  const _Float16* src = H16A;
  _Float16*       dst = H16B;
  for (int step = 0; step < 4; step++) {
    lstm_step_kernel<<<grid, 256, 0, stream>>>(src, dst, C32, XW, WHH16,
                                               pos_tem, neg_tem, step);
    const _Float16* t = dst; dst = (_Float16*)src; src = t;  // ping-pong h
  }
  // After 4 swaps the final hidden state is back in H16A (== src).
  score_kernel<<<M_ALL, 256, 0, stream>>>(src, ent_emb, pos_h, pos_t,
                                          neg_h, neg_t, (float*)d_out);
}
